// HybridTrojanNetPyG_82798379532726
// MI455X (gfx1250) — compile-verified, matrix operand order checked by reading, not run on hardware
//
#include <hip/hip_runtime.h>

#define Nn 4096
#define Ee 262144
#define ENe (Ee + Nn)

typedef __attribute__((ext_vector_type(16))) _Float16 v16h;
typedef __attribute__((ext_vector_type(8)))  float    v8f;
typedef __attribute__((ext_vector_type(8)))  unsigned v8u;

// ---------- helpers ----------
static __device__ __forceinline__ unsigned fenc(float f) {
  unsigned u = __float_as_uint(f);
  return (u & 0x80000000u) ? ~u : (u | 0x80000000u);
}
static __device__ __forceinline__ float fdec(unsigned u) {
  u = (u & 0x80000000u) ? (u & 0x7FFFFFFFu) : ~u;
  return __uint_as_float(u);
}
static __device__ __forceinline__ unsigned pack2(_Float16 a, _Float16 b) {
  return (unsigned)__builtin_bit_cast(unsigned short, a) |
         ((unsigned)__builtin_bit_cast(unsigned short, b) << 16);
}
// K index for pair p (0..7) in a 16-bit A/B fragment (ISA 7.12.2 layout)
static __device__ __forceinline__ int kpair(int p, int hlf) {
  return ((p < 4) ? 2 * p : 16 + 2 * (p - 4)) + hlf * 8;
}

// ---------- small utility kernels ----------
__global__ void k_fill_f32(float* p, int n, float v) {
  int i = blockIdx.x * blockDim.x + threadIdx.x;
  if (i < n) p[i] = v;
}
__global__ void k_fill_u32(unsigned* p, int n, unsigned v) {
  int i = blockIdx.x * blockDim.x + threadIdx.x;
  if (i < n) p[i] = v;
}
__global__ void k_f32_to_f16(const float* a, _Float16* b, int n) {
  int i = blockIdx.x * blockDim.x + threadIdx.x;
  if (i < n) b[i] = (_Float16)a[i];
}
// Wt[n*K+k] = (f16) W[k*Nc+n]   (transpose + convert weights)
__global__ void k_wconv(const float* W, _Float16* Wt, int K, int Nc) {
  int i = blockIdx.x * blockDim.x + threadIdx.x;
  if (i >= K * Nc) return;
  int n = i / K, k = i % K;
  Wt[i] = (_Float16)W[k * Nc + n];
}

// ---------- WMMA GEMM: C[M,Nc] = act(A[M,K] @ B[K,Nc] + bias) * rowscale ----------
// A: f16 row-major. Bt: f16, B transposed (Bt[n*K+k] = B[k][n]). One 16x16 tile per wave.
__global__ void k_gemm(const _Float16* __restrict__ A, const _Float16* __restrict__ Bt,
                       const float* bias, const float* rowscale, float* C,
                       int M, int K, int Nc, int act) {
  int wave = (blockIdx.x * blockDim.x + threadIdx.x) >> 5;
  int lane = threadIdx.x & 31;
  int tilesN = Nc >> 4;
  int tm = (wave / tilesN) << 4;
  int tn = (wave % tilesN) << 4;
  if (tm >= M) return;                 // uniform per-wave exit, EXEC stays full
  int hlf = lane >> 4, mn = lane & 15;
  const _Float16* arow = A + (size_t)(tm + mn) * K;
  const _Float16* brow = Bt + (size_t)(tn + mn) * K;
  v8f acc = {};
  for (int k0 = 0; k0 < K; k0 += 32) {
    if (k0 + 32 < K) {                 // prefetch next K tile (global_prefetch_b8)
      __builtin_prefetch(arow + k0 + 32, 0, 0);
      __builtin_prefetch(brow + k0 + 32, 0, 0);
    }
    v8u au, bu;
#pragma unroll
    for (int p = 0; p < 8; ++p) {
      int kk = k0 + kpair(p, hlf);
      au[p] = *(const unsigned*)(arow + kk);
      bu[p] = *(const unsigned*)(brow + kk);
    }
    acc = __builtin_amdgcn_wmma_f32_16x16x32_f16(
        false, __builtin_bit_cast(v16h, au),
        false, __builtin_bit_cast(v16h, bu),
        (short)0, acc, false, false);
  }
  int n = tn + mn;
  float bvv = bias ? bias[n] : 0.f;
#pragma unroll
  for (int r = 0; r < 8; ++r) {
    int m = tm + r + hlf * 8;
    float v = acc[r] + bvv;
    if (act) v = fmaxf(v, 0.f);
    if (rowscale) v *= rowscale[m];
    C[(size_t)m * Nc + n] = v;
  }
}

// ---------- node init: embedding + tiny MLPs (K=5, below WMMA granularity) ----------
__global__ void k_node_init(const int* gid, const float* nf, const float* emb,
                            const float* npW1, const float* npb1,
                            const float* npW2, const float* npb2,
                            const float* rgW1, const float* rgb1,
                            const float* rgW2, const float* rgb2,
                            _Float16* cat, float* riskp1) {
  int n = blockIdx.x, t = threadIdx.x;
  __shared__ float f[5];
  __shared__ float h1[32];
  __shared__ float red[128];
  if (t < 5) f[t] = nf[n * 5 + t];
  __syncthreads();
  if (t < 32) {
    float a = npb1[t];
#pragma unroll
    for (int g = 0; g < 5; ++g) a += f[g] * npW1[g * 32 + t];
    h1[t] = fmaxf(a, 0.f);
  }
  float a = rgb1[t];
#pragma unroll
  for (int g = 0; g < 5; ++g) a += f[g] * rgW1[g * 128 + t];
  a = fmaxf(a, 0.f);
  red[t] = a * rgW2[t];
  __syncthreads();
  for (int o = 64; o > 0; o >>= 1) {
    if (t < o) red[t] += red[t + o];
    __syncthreads();
  }
  if (t == 0) {
    float z = red[0] + rgb2[0];
    riskp1[n] = 1.f + 1.f / (1.f + __expf(-z));
  }
  if (t < 32) {
    float b2 = npb2[t];
    for (int j = 0; j < 32; ++j) b2 += h1[j] * npW2[j * 32 + t];
    cat[(size_t)n * 64 + 32 + t] = (_Float16)b2;
    cat[(size_t)n * 64 + t] = (_Float16)emb[gid[n] * 32 + t];
  }
}

// ---------- GIN scatter-add + combine ----------
__global__ void k_gin_scatter(const float* x, const int* src, const int* dst, float* agg) {
  int t = blockIdx.x * blockDim.x + threadIdx.x;
  if (t >= Ee * 32) return;
  int e = t >> 5, c4 = (t & 31) * 4;
  int s = src[e], d = dst[e];
#pragma unroll
  for (int j = 0; j < 4; ++j)
    atomicAdd(&agg[(size_t)d * 128 + c4 + j], x[(size_t)s * 128 + c4 + j]);
}
__global__ void k_gin_combine(const float* x, const float* agg, const float* eps, _Float16* o) {
  int i = blockIdx.x * blockDim.x + threadIdx.x;
  if (i < Nn * 128) o[i] = (_Float16)((1.f + eps[0]) * x[i] + agg[i]);
}

// ---------- GATv2 ----------
__global__ void k_deg_loop(const int* dstp, const float* eattr, float* deg, float* loopa) {
  int e = blockIdx.x * blockDim.x + threadIdx.x;
  if (e >= Ee) return;
  int d = dstp[e];
  atomicAdd(&deg[d], 1.f);
  atomicAdd(&loopa[d * 3 + 0], eattr[e * 3 + 0]);
  atomicAdd(&loopa[d * 3 + 1], eattr[e * 3 + 1]);
  atomicAdd(&loopa[d * 3 + 2], eattr[e * 3 + 2]);
}
__global__ void k_loopdiv(float* loopa, const float* deg) {
  int n = blockIdx.x * blockDim.x + threadIdx.x;
  if (n >= Nn) return;
  float dg = fmaxf(deg[n], 1.f);
  loopa[n * 3 + 0] /= dg; loopa[n * 3 + 1] /= dg; loopa[n * 3 + 2] /= dg;
}
__global__ void k_gat_logit(const float* xl, const float* xr, const int* src, const int* dst,
                            const float* eattr, const float* loopa,
                            const float* We, const float* att,
                            float* logit, unsigned* segmax) {
  int i = blockIdx.x * blockDim.x + threadIdx.x;
  if (i >= ENe) return;
  int s, d; float e0, e1, e2;
  if (i < Ee) { s = src[i]; d = dst[i]; e0 = eattr[i*3]; e1 = eattr[i*3+1]; e2 = eattr[i*3+2]; }
  else { s = d = i - Ee; e0 = loopa[s*3]; e1 = loopa[s*3+1]; e2 = loopa[s*3+2]; }
  const float* xls = xl + (size_t)s * 128;
  const float* xrd = xr + (size_t)d * 128;
#pragma unroll
  for (int hd = 0; hd < 4; ++hd) {
    float acc = 0.f;
    for (int od = 0; od < 32; ++od) {
      int c = hd * 32 + od;
      float m = xls[c] + xrd[c] + e0 * We[c] + e1 * We[128 + c] + e2 * We[256 + c];
      m = (m > 0.f) ? m : 0.2f * m;
      acc += m * att[c];
    }
    logit[(size_t)i * 4 + hd] = acc;
    atomicMax(&segmax[d * 4 + hd], fenc(acc));
  }
}
__global__ void k_gat_expsum(const int* dst, float* logit, const unsigned* segmax, float* segsum) {
  int i = blockIdx.x * blockDim.x + threadIdx.x;
  if (i >= ENe) return;
  int d = (i < Ee) ? dst[i] : i - Ee;
#pragma unroll
  for (int hd = 0; hd < 4; ++hd) {
    float m = fdec(segmax[d * 4 + hd]);
    float ex = __expf(logit[(size_t)i * 4 + hd] - m);
    logit[(size_t)i * 4 + hd] = ex;
    atomicAdd(&segsum[d * 4 + hd], ex);
  }
}
__global__ void k_gat_scatter(const int* src, const int* dst, const float* logit,
                              const float* segsum, const float* xl, float* outb) {
  long t = blockIdx.x * (long)blockDim.x + threadIdx.x;
  if (t >= (long)ENe * 32) return;
  int i = (int)(t >> 5), c4 = ((int)t & 31) * 4;
  int s, d;
  if (i < Ee) { s = src[i]; d = dst[i]; } else { s = d = i - Ee; }
#pragma unroll
  for (int j = 0; j < 4; ++j) {
    int c = c4 + j, hd = c >> 5;
    float alpha = logit[(size_t)i * 4 + hd] / segsum[d * 4 + hd];
    atomicAdd(&outb[(size_t)d * 128 + c], alpha * xl[(size_t)s * 128 + c]);
  }
}

// ---------- LayerNorm of x_local + 0.5(xf+bf) + 0.5(xr+br), wave per node ----------
__global__ void k_ln_dir(const float* xloc, const float* xf, const float* xr,
                         const float* bf, const float* br,
                         const float* g, const float* b, float* outb) {
  int wave = (blockIdx.x * blockDim.x + threadIdx.x) >> 5;
  int lane = threadIdx.x & 31;
  if (wave >= Nn) return;
  size_t base = (size_t)wave * 128;
  float v[4]; float s = 0.f;
#pragma unroll
  for (int j = 0; j < 4; ++j) {
    int c = lane * 4 + j;
    float t = xloc[base + c] + 0.5f * (xf[base + c] + bf[c]) + 0.5f * (xr[base + c] + br[c]);
    v[j] = t; s += t;
  }
  s += __shfl_xor(s, 1); s += __shfl_xor(s, 2); s += __shfl_xor(s, 4);
  s += __shfl_xor(s, 8); s += __shfl_xor(s, 16);
  float mean = s * (1.f / 128.f);
  float var = 0.f;
#pragma unroll
  for (int j = 0; j < 4; ++j) { float dv = v[j] - mean; var += dv * dv; }
  var += __shfl_xor(var, 1); var += __shfl_xor(var, 2); var += __shfl_xor(var, 4);
  var += __shfl_xor(var, 8); var += __shfl_xor(var, 16);
  var *= (1.f / 128.f);
  float inv = rsqrtf(var + 1e-5f);
#pragma unroll
  for (int j = 0; j < 4; ++j) {
    int c = lane * 4 + j;
    outb[base + c] = g[c] * (v[j] - mean) * inv + b[c];
  }
}

// ---------- BatchNorm(a + c2) over nodes, block per feature ----------
__global__ void k_bn_add(const float* a, const float* c2, const float* g, const float* bb,
                         float* outb) {
  int f = blockIdx.x;
  __shared__ float r1[256], r2[256];
  float s = 0.f, s2 = 0.f;
  for (int n = threadIdx.x; n < Nn; n += 256) {
    float v = a[(size_t)n * 128 + f] + c2[(size_t)n * 128 + f];
    s += v; s2 += v * v;
  }
  r1[threadIdx.x] = s; r2[threadIdx.x] = s2;
  __syncthreads();
  for (int o = 128; o > 0; o >>= 1) {
    if (threadIdx.x < o) { r1[threadIdx.x] += r1[threadIdx.x + o]; r2[threadIdx.x] += r2[threadIdx.x + o]; }
    __syncthreads();
  }
  float mean = r1[0] * (1.f / Nn);
  float var = r2[0] * (1.f / Nn) - mean * mean;
  float inv = rsqrtf(var + 1e-5f);
  float gg = g[f], bv = bb[f];
  for (int n = threadIdx.x; n < Nn; n += 256) {
    float v = a[(size_t)n * 128 + f] + c2[(size_t)n * 128 + f];
    outb[(size_t)n * 128 + f] = gg * (v - mean) * inv + bv;
  }
}

// ---------- flash attention ----------
// Wave per (head, 16-row query tile); the 4 waves of a block share one head and
// stream identical K/V tiles, so each 32x32 f16 K and V tile is staged into LDS
// ONCE per block with global_load_async_to_lds_b128 (ASYNCcnt path), then all
// waves read their WMMA fragments from LDS (ds_load co-issues with v_wmma).
__global__ void k_mha_flash(const _Float16* __restrict__ qh, const _Float16* __restrict__ kh,
                            const _Float16* __restrict__ vh, float* oall) {
  __shared__ __align__(16) _Float16 ldsK[32 * 32];   // [key][dim] 2KB
  __shared__ __align__(16) _Float16 ldsV[32 * 32];   // [key][dim] 2KB
  __shared__ __align__(8)  _Float16 ldsP[4][16 * 32]; // per-wave P staging
  int wid = threadIdx.x >> 5;
  int lane = threadIdx.x & 31;
  int wave = blockIdx.x * 4 + wid;
  int qt = wave & 255;
  int hd = wave >> 8;            // block-uniform (4 aligned waves within a 256 group)
  int hlf = lane >> 4, mn = lane & 15;
  int col0 = hd * 32;
  const float scale = 0.17677669529663687f; // 1/sqrt(32)
  // staging role: each wave moves 8 rows of K and 8 rows of V, 16B per lane
  const int srow = wid * 8 + (lane >> 2);   // local key row 0..31
  const int scol = (lane & 3) * 8;          // halfs within 32-dim row
  unsigned ldsKo = (unsigned)(unsigned long long)(const void*)(ldsK + srow * 32 + scol);
  unsigned ldsVo = (unsigned)(unsigned long long)(const void*)(ldsV + srow * 32 + scol);
  // Q fragment: A 16x32, rows qt*16.., dims col0..col0+31
  v8u aqu;
  const _Float16* qrow = qh + (size_t)(qt * 16 + mn) * 128 + col0;
#pragma unroll
  for (int p = 0; p < 8; ++p) aqu[p] = *(const unsigned*)(qrow + kpair(p, hlf));
  v16h aq = __builtin_bit_cast(v16h, aqu);
  float mrow[8], lrow[8];
#pragma unroll
  for (int r = 0; r < 8; ++r) { mrow[r] = -1e30f; lrow[r] = 0.f; }
  v8f o0 = {}, o1 = {};
  _Float16* pt = ldsP[wid];
  for (int kt = 0; kt < Nn; kt += 32) {
    // async-stage this block's K/V tile into LDS
    {
      unsigned long long gK =
          (unsigned long long)(const void*)(kh + (size_t)(kt + srow) * 128 + col0 + scol);
      unsigned long long gV =
          (unsigned long long)(const void*)(vh + (size_t)(kt + srow) * 128 + col0 + scol);
      asm volatile("global_load_async_to_lds_b128 %0, %1, off"
                   :: "v"(ldsKo), "v"(gK) : "memory");
      asm volatile("global_load_async_to_lds_b128 %0, %1, off"
                   :: "v"(ldsVo), "v"(gV) : "memory");
      asm volatile("s_wait_asynccnt 0x0" ::: "memory");
    }
    __syncthreads();
    // K^T fragments from LDS (keys kt..kt+15 and kt+16..kt+31)
    v8u b0u, b1u;
#pragma unroll
    for (int p = 0; p < 8; ++p) {
      int kk = kpair(p, hlf);
      b0u[p] = *(const unsigned*)(ldsK + mn * 32 + kk);
      b1u[p] = *(const unsigned*)(ldsK + (16 + mn) * 32 + kk);
    }
    v8f zacc = {};
    v8f s0 = __builtin_amdgcn_wmma_f32_16x16x32_f16(false, aq, false,
               __builtin_bit_cast(v16h, b0u), (short)0, zacc, false, false);
    v8f s1 = __builtin_amdgcn_wmma_f32_16x16x32_f16(false, aq, false,
               __builtin_bit_cast(v16h, b1u), (short)0, zacc, false, false);
    // online softmax; row m = r + hlf*8 lives in 16-lane group, vgpr r
#pragma unroll
    for (int r = 0; r < 8; ++r) {
      float a0 = s0[r] * scale, a1 = s1[r] * scale;
      float rm = fmaxf(a0, a1);
      rm = fmaxf(rm, __shfl_xor(rm, 1)); rm = fmaxf(rm, __shfl_xor(rm, 2));
      rm = fmaxf(rm, __shfl_xor(rm, 4)); rm = fmaxf(rm, __shfl_xor(rm, 8));
      float mnew = fmaxf(mrow[r], rm);
      float corr = __expf(mrow[r] - mnew);
      float e0 = __expf(a0 - mnew), e1 = __expf(a1 - mnew);
      float rs = e0 + e1;
      rs += __shfl_xor(rs, 1); rs += __shfl_xor(rs, 2);
      rs += __shfl_xor(rs, 4); rs += __shfl_xor(rs, 8);
      lrow[r] = lrow[r] * corr + rs;
      mrow[r] = mnew;
      o0[r] *= corr; o1[r] *= corr;
      int m = r + hlf * 8;
      pt[m * 32 + mn] = (_Float16)e0;        // P row-major in LDS
      pt[m * 32 + 16 + mn] = (_Float16)e1;
    }
    // reload P as A fragment (C-layout -> A-layout transpose via LDS)
    v8u apu;
#pragma unroll
    for (int p = 0; p < 8; ++p)
      apu[p] = *(const unsigned*)(pt + mn * 32 + kpair(p, hlf));
    v16h ap = __builtin_bit_cast(v16h, apu);
    // V fragments from LDS: B[k=key][n=dim], dims split 0..15 / 16..31
    v8u bv0u, bv1u;
#pragma unroll
    for (int p = 0; p < 8; ++p) {
      int kk = kpair(p, hlf);
      bv0u[p] = pack2(ldsV[kk * 32 + mn], ldsV[(kk + 1) * 32 + mn]);
      bv1u[p] = pack2(ldsV[kk * 32 + 16 + mn], ldsV[(kk + 1) * 32 + 16 + mn]);
    }
    o0 = __builtin_amdgcn_wmma_f32_16x16x32_f16(false, ap, false,
           __builtin_bit_cast(v16h, bv0u), (short)0, o0, false, false);
    o1 = __builtin_amdgcn_wmma_f32_16x16x32_f16(false, ap, false,
           __builtin_bit_cast(v16h, bv1u), (short)0, o1, false, false);
    __syncthreads();  // K/V tile consumed before next iteration overwrites it
  }
#pragma unroll
  for (int r = 0; r < 8; ++r) {
    int m = qt * 16 + r + hlf * 8;
    float inv = 1.f / lrow[r];
    oall[(size_t)m * 128 + col0 + mn] = o0[r] * inv;
    oall[(size_t)m * 128 + col0 + 16 + mn] = o1[r] * inv;
  }
}

// ---------- fuse concat + classifier tail ----------
__global__ void k_cat3(const float* a, const float* b, const float* c, _Float16* o) {
  int i = blockIdx.x * blockDim.x + threadIdx.x;
  if (i >= Nn * 384) return;
  int n = i / 384, col = i % 384;
  float v = (col < 128) ? a[(size_t)n * 128 + col]
          : (col < 256) ? b[(size_t)n * 128 + col - 128]
                        : c[(size_t)n * 128 + col - 256];
  o[i] = (_Float16)v;
}
__global__ void k_cls_final(const float* c1, const float* W2, const float* b2, float* out) {
  int wave = (blockIdx.x * blockDim.x + threadIdx.x) >> 5;
  int lane = threadIdx.x & 31;
  if (wave >= Nn) return;
  float s = c1[(size_t)wave * 64 + lane] * W2[lane] +
            c1[(size_t)wave * 64 + 32 + lane] * W2[32 + lane];
  s += __shfl_xor(s, 1); s += __shfl_xor(s, 2); s += __shfl_xor(s, 4);
  s += __shfl_xor(s, 8); s += __shfl_xor(s, 16);
  if (lane == 0) out[wave] = s + b2[0];
}

// ---------- host ----------
extern "C" void kernel_launch(void* const* d_in, const int* in_sizes, int n_in,
                              void* d_out, int out_size, void* d_ws, size_t ws_size,
                              hipStream_t stream) {
  (void)in_sizes; (void)n_in; (void)out_size; (void)ws_size;
  const int*   gid   = (const int*)d_in[0];
  const float* nfeat = (const float*)d_in[1];
  const int*   ei    = (const int*)d_in[2];
  const float* eattr = (const float*)d_in[3];
  const float* emb   = (const float*)d_in[4];
  const float* npW1  = (const float*)d_in[5];
  const float* npb1  = (const float*)d_in[6];
  const float* npW2  = (const float*)d_in[7];
  const float* npb2  = (const float*)d_in[8];
  const float* inW   = (const float*)d_in[9];
  const float* inb   = (const float*)d_in[10];
  const float* rgW1  = (const float*)d_in[11];
  const float* rgb1  = (const float*)d_in[12];
  const float* rgW2  = (const float*)d_in[13];
  const float* rgb2  = (const float*)d_in[14];
  const float* geps  = (const float*)d_in[15];
  const float* ginW1 = (const float*)d_in[16];
  const float* ginb1 = (const float*)d_in[17];
  const float* ginW2 = (const float*)d_in[18];
  const float* ginb2 = (const float*)d_in[19];
  const float* gfWl  = (const float*)d_in[20];
  const float* gfbl  = (const float*)d_in[21];
  const float* gfWr  = (const float*)d_in[22];
  const float* gfbr  = (const float*)d_in[23];
  const float* gfWe  = (const float*)d_in[24];
  const float* gfatt = (const float*)d_in[25];
  const float* gfb   = (const float*)d_in[26];
  const float* grWl  = (const float*)d_in[27];
  const float* grbl  = (const float*)d_in[28];
  const float* grWr  = (const float*)d_in[29];
  const float* grbr  = (const float*)d_in[30];
  const float* grWe  = (const float*)d_in[31];
  const float* gratt = (const float*)d_in[32];
  const float* grb   = (const float*)d_in[33];
  const float* dng   = (const float*)d_in[34];
  const float* dnb   = (const float*)d_in[35];
  const float* Wq    = (const float*)d_in[36];
  const float* Wk    = (const float*)d_in[37];
  const float* Wv    = (const float*)d_in[38];
  const float* Wo    = (const float*)d_in[39];
  const float* bq    = (const float*)d_in[40];
  const float* bk    = (const float*)d_in[41];
  const float* bvp   = (const float*)d_in[42];
  const float* bo    = (const float*)d_in[43];
  const float* bn2g  = (const float*)d_in[44];
  const float* bn2b  = (const float*)d_in[45];
  const float* mW1   = (const float*)d_in[46];
  const float* mb1   = (const float*)d_in[47];
  const float* mW2   = (const float*)d_in[48];
  const float* mb2   = (const float*)d_in[49];
  const float* bn3g  = (const float*)d_in[50];
  const float* bn3b  = (const float*)d_in[51];
  const float* fW    = (const float*)d_in[52];
  const float* fb    = (const float*)d_in[53];
  const float* cW1   = (const float*)d_in[54];
  const float* cb1   = (const float*)d_in[55];
  const float* cW2   = (const float*)d_in[56];
  const float* cb2   = (const float*)d_in[57];
  float* out = (float*)d_out;
  const int* srcp = ei;
  const int* dstp = ei + Ee;

  char* wsp = (char*)d_ws;
  size_t off = 0;
  auto alloc = [&](size_t bytes) -> void* {
    void* p = wsp + off;
    off = (off + bytes + 255) & ~((size_t)255);
    return p;
  };
  // f16 transposed weights
  _Float16* inWt   = (_Float16*)alloc(64 * 128 * 2);
  _Float16* ginW1t = (_Float16*)alloc(128 * 128 * 2);
  _Float16* ginW2t = (_Float16*)alloc(128 * 128 * 2);
  _Float16* gfWlt  = (_Float16*)alloc(128 * 128 * 2);
  _Float16* gfWrt  = (_Float16*)alloc(128 * 128 * 2);
  _Float16* grWlt  = (_Float16*)alloc(128 * 128 * 2);
  _Float16* grWrt  = (_Float16*)alloc(128 * 128 * 2);
  _Float16* Wqt    = (_Float16*)alloc(128 * 128 * 2);
  _Float16* Wkt    = (_Float16*)alloc(128 * 128 * 2);
  _Float16* Wvt    = (_Float16*)alloc(128 * 128 * 2);
  _Float16* Wot    = (_Float16*)alloc(128 * 128 * 2);
  _Float16* mW1t   = (_Float16*)alloc(128 * 256 * 2);
  _Float16* mW2t   = (_Float16*)alloc(256 * 128 * 2);
  _Float16* fWt    = (_Float16*)alloc(384 * 128 * 2);
  _Float16* cW1t   = (_Float16*)alloc(128 * 64 * 2);
  // f16 activations
  _Float16* cat64h = (_Float16*)alloc((size_t)Nn * 64 * 2);
  _Float16* ginh   = (_Float16*)alloc((size_t)Nn * 128 * 2);
  _Float16* t1h    = (_Float16*)alloc((size_t)Nn * 128 * 2);
  _Float16* xloch  = (_Float16*)alloc((size_t)Nn * 128 * 2);
  _Float16* xdirh  = (_Float16*)alloc((size_t)Nn * 128 * 2);
  _Float16* qhb    = (_Float16*)alloc((size_t)Nn * 128 * 2);
  _Float16* khb    = (_Float16*)alloc((size_t)Nn * 128 * 2);
  _Float16* vhb    = (_Float16*)alloc((size_t)Nn * 128 * 2);
  _Float16* oallh  = (_Float16*)alloc((size_t)Nn * 128 * 2);
  _Float16* hattnh = (_Float16*)alloc((size_t)Nn * 128 * 2);
  _Float16* mhidh  = (_Float16*)alloc((size_t)Nn * 256 * 2);
  _Float16* cat3h  = (_Float16*)alloc((size_t)Nn * 384 * 2);
  _Float16* xfush  = (_Float16*)alloc((size_t)Nn * 128 * 2);
  // f32
  float* riskp1 = (float*)alloc((size_t)Nn * 4);
  float* x      = (float*)alloc((size_t)Nn * 128 * 4);
  float* agg    = (float*)alloc((size_t)Nn * 128 * 4);
  float* t1     = (float*)alloc((size_t)Nn * 128 * 4);
  float* xloc   = (float*)alloc((size_t)Nn * 128 * 4);
  float* xl     = (float*)alloc((size_t)Nn * 128 * 4);
  float* xr     = (float*)alloc((size_t)Nn * 128 * 4);
  float* outf   = (float*)alloc((size_t)Nn * 128 * 4);
  float* outr   = (float*)alloc((size_t)Nn * 128 * 4);
  float* degf   = (float*)alloc((size_t)Nn * 4);
  float* loopf  = (float*)alloc((size_t)Nn * 3 * 4);
  float* degr   = (float*)alloc((size_t)Nn * 4);
  float* loopr  = (float*)alloc((size_t)Nn * 3 * 4);
  float* logitb = (float*)alloc((size_t)ENe * 4 * 4);
  unsigned* segmax = (unsigned*)alloc((size_t)Nn * 4 * 4);
  float* segsum = (float*)alloc((size_t)Nn * 4 * 4);
  float* xdir   = (float*)alloc((size_t)Nn * 128 * 4);
  float* qf     = (float*)alloc((size_t)Nn * 128 * 4);
  float* kf     = (float*)alloc((size_t)Nn * 128 * 4);
  float* vf     = (float*)alloc((size_t)Nn * 128 * 4);
  float* oall   = (float*)alloc((size_t)Nn * 128 * 4);
  float* attn   = (float*)alloc((size_t)Nn * 128 * 4);
  float* hattn  = (float*)alloc((size_t)Nn * 128 * 4);
  float* mhid   = (float*)alloc((size_t)Nn * 256 * 4);
  float* mout   = (float*)alloc((size_t)Nn * 128 * 4);
  float* xglob  = (float*)alloc((size_t)Nn * 128 * 4);
  float* xfus   = (float*)alloc((size_t)Nn * 128 * 4);
  float* c1     = (float*)alloc((size_t)Nn * 64 * 4);

  auto wc = [&](const float* W, _Float16* Wt, int K, int Ncol) {
    int cnt = K * Ncol;
    k_wconv<<<(cnt + 255) / 256, 256, 0, stream>>>(W, Wt, K, Ncol);
  };
  auto gemm = [&](const _Float16* A, const _Float16* Bt, const float* bias,
                  const float* rs, float* C, int M, int K, int Ncol, int act) {
    int waves = (M / 16) * (Ncol / 16);
    int thr = waves * 32;
    k_gemm<<<(thr + 255) / 256, 256, 0, stream>>>(A, Bt, bias, rs, C, M, K, Ncol, act);
  };
  auto cvt = [&](const float* a, _Float16* b, int cnt) {
    k_f32_to_f16<<<(cnt + 255) / 256, 256, 0, stream>>>(a, b, cnt);
  };
  auto fillf = [&](float* p, int cnt) {
    k_fill_f32<<<(cnt + 255) / 256, 256, 0, stream>>>(p, cnt, 0.f);
  };
  auto fillu = [&](unsigned* p, int cnt) {
    k_fill_u32<<<(cnt + 255) / 256, 256, 0, stream>>>(p, cnt, 0u);
  };

  // weight prep
  wc(inW, inWt, 64, 128);
  wc(ginW1, ginW1t, 128, 128); wc(ginW2, ginW2t, 128, 128);
  wc(gfWl, gfWlt, 128, 128);   wc(gfWr, gfWrt, 128, 128);
  wc(grWl, grWlt, 128, 128);   wc(grWr, grWrt, 128, 128);
  wc(Wq, Wqt, 128, 128); wc(Wk, Wkt, 128, 128);
  wc(Wv, Wvt, 128, 128); wc(Wo, Wot, 128, 128);
  wc(mW1, mW1t, 128, 256); wc(mW2, mW2t, 256, 128);
  wc(fW, fWt, 384, 128);   wc(cW1, cW1t, 128, 64);

  // node features -> x (fused relu + (1+risk) row-scale)
  k_node_init<<<Nn, 128, 0, stream>>>(gid, nfeat, emb, npW1, npb1, npW2, npb2,
                                      rgW1, rgb1, rgW2, rgb2, cat64h, riskp1);
  gemm(cat64h, inWt, inb, riskp1, x, Nn, 64, 128, 1);

  // GIN
  fillf(agg, Nn * 128);
  k_gin_scatter<<<(Ee * 32 + 255) / 256, 256, 0, stream>>>(x, srcp, dstp, agg);
  k_gin_combine<<<(Nn * 128 + 255) / 256, 256, 0, stream>>>(x, agg, geps, ginh);
  gemm(ginh, ginW1t, ginb1, nullptr, t1, Nn, 128, 128, 1);
  cvt(t1, t1h, Nn * 128);
  gemm(t1h, ginW2t, ginb2, nullptr, xloc, Nn, 128, 128, 0);
  cvt(xloc, xloch, Nn * 128);

  // GATv2 (both directions)
  auto gat_dir = [&](const int* s_, const int* d_, const float* We, const float* att,
                     float* deg, float* loopa, float* outbuf) {
    fillf(deg, Nn); fillf(loopa, Nn * 3);
    k_deg_loop<<<(Ee + 255) / 256, 256, 0, stream>>>(d_, eattr, deg, loopa);
    k_loopdiv<<<(Nn + 255) / 256, 256, 0, stream>>>(loopa, deg);
    fillu(segmax, Nn * 4); fillf(segsum, Nn * 4);
    k_gat_logit<<<(ENe + 255) / 256, 256, 0, stream>>>(xl, xr, s_, d_, eattr, loopa,
                                                       We, att, logitb, segmax);
    k_gat_expsum<<<(ENe + 255) / 256, 256, 0, stream>>>(d_, logitb, segmax, segsum);
    fillf(outbuf, Nn * 128);
    long tot = (long)ENe * 32;
    k_gat_scatter<<<(int)((tot + 255) / 256), 256, 0, stream>>>(s_, d_, logitb, segsum, xl, outbuf);
  };
  gemm(xloch, gfWlt, gfbl, nullptr, xl, Nn, 128, 128, 0);
  gemm(xloch, gfWrt, gfbr, nullptr, xr, Nn, 128, 128, 0);
  gat_dir(srcp, dstp, gfWe, gfatt, degf, loopf, outf);
  gemm(xloch, grWlt, grbl, nullptr, xl, Nn, 128, 128, 0);
  gemm(xloch, grWrt, grbr, nullptr, xr, Nn, 128, 128, 0);
  gat_dir(dstp, srcp, grWe, gratt, degr, loopr, outr);

  k_ln_dir<<<Nn / 8, 256, 0, stream>>>(xloc, outf, outr, gfb, grb, dng, dnb, xdir);
  cvt(xdir, xdirh, Nn * 128);

  // global attention (GPSConv)
  gemm(xdirh, Wqt, bq, nullptr, qf, Nn, 128, 128, 0);
  gemm(xdirh, Wkt, bk, nullptr, kf, Nn, 128, 128, 0);
  gemm(xdirh, Wvt, bvp, nullptr, vf, Nn, 128, 128, 0);
  cvt(qf, qhb, Nn * 128); cvt(kf, khb, Nn * 128); cvt(vf, vhb, Nn * 128);
  k_mha_flash<<<256, 128, 0, stream>>>(qhb, khb, vhb, oall);
  cvt(oall, oallh, Nn * 128);
  gemm(oallh, Wot, bo, nullptr, attn, Nn, 128, 128, 0);
  k_bn_add<<<128, 256, 0, stream>>>(attn, xdir, bn2g, bn2b, hattn);
  cvt(hattn, hattnh, Nn * 128);
  gemm(hattnh, mW1t, mb1, nullptr, mhid, Nn, 128, 256, 1);
  cvt(mhid, mhidh, Nn * 256);
  gemm(mhidh, mW2t, mb2, nullptr, mout, Nn, 256, 128, 0);
  k_bn_add<<<128, 256, 0, stream>>>(mout, hattn, bn3g, bn3b, xglob);

  // fuse + classifier
  k_cat3<<<(Nn * 384 + 255) / 256, 256, 0, stream>>>(xloc, xdir, xglob, cat3h);
  gemm(cat3h, fWt, fb, nullptr, xfus, Nn, 384, 128, 1);
  cvt(xfus, xfush, Nn * 128);
  gemm(xfush, cW1t, cb1, nullptr, c1, Nn, 128, 64, 1);
  k_cls_final<<<Nn * 32 / 256, 256, 0, stream>>>(c1, cW2, cb2, out);
}
